// MHLLM_19310172963167
// MI455X (gfx1250) — matrix operand order repaired and checked by reading, not compiled, not run on hardware
//
#include <hip/hip_runtime.h>
#include <stdint.h>

// ---------------------------------------------------------------------------
// Problem dims (fixed by the reference)
// ---------------------------------------------------------------------------
#define B_DIM   4
#define T_DIM   2048
#define D_DIM   1024
#define V_DIM   32000
#define M_DIM   (B_DIM * T_DIM)     // 8192 rows of logits
#define CHUNK_T 128
#define NCHUNK  (T_DIM / CHUNK_T)   // 16

// GEMM tiling
#define BLK_M   128
#define BLK_N   128
#define KC      32                  // K per stage chunk
#define NKC     (D_DIM / KC)        // 32 chunks
#define LDS_PITCH 80                // 64B of chunk data + 16B pad (bank-conflict-free)
#define PLANE_BYTES (128 * LDS_PITCH)      // 10240 B : one bf16 plane (128 rows x 32 k)
#define BUF_BYTES   (4 * PLANE_BYTES)      // A_hi, A_lo, B_hi, B_lo

typedef __attribute__((ext_vector_type(16))) __bf16 v16bf;
typedef __attribute__((ext_vector_type(8)))  __bf16 v8bf;
typedef __attribute__((ext_vector_type(8)))  float  v8f;

union AFrag { v16bf v; v8bf h[2]; };

static __device__ __forceinline__ unsigned short bf_bits(__bf16 b) {
  union { __bf16 b; unsigned short u; } c; c.b = b; return c.u;
}

// fp32 -> bf16 hi + bf16 lo (error-compensated split: x ~= hi + lo)
static __device__ __forceinline__ void split_f32(float x, __bf16 &hi, __bf16 &lo) {
  hi = (__bf16)x;
  lo = (__bf16)(x - (float)hi);
}

// CDNA5 async DMA: per-lane 16B global -> LDS, tracked by ASYNCcnt (ISA 15.18.3 op 98)
static __device__ __forceinline__ void async_ld_b128(uint32_t lds_off, uint64_t sbase,
                                                     uint32_t voff) {
  asm volatile("global_load_async_to_lds_b128 %0, %1, %2"
               :: "v"(lds_off), "v"(voff), "s"(sbase) : "memory");
}
static __device__ __forceinline__ void wait_asynccnt0() {
  asm volatile("s_wait_asynccnt 0x0" ::: "memory");
}

// ---------------------------------------------------------------------------
// Stage 1a: per-chunk sums of gathered embeddings (breaks the T-long chain)
// ---------------------------------------------------------------------------
__global__ __launch_bounds__(256) void MHLLM_chunk_sums(
    const int* __restrict__ x, const float* __restrict__ emb,
    float* __restrict__ csum)
{
  const int b = blockIdx.x / NCHUNK;
  const int c = blockIdx.x % NCHUNK;
  const int d = blockIdx.y * 256 + threadIdx.x;
  const int* xr = x + b * T_DIM + c * CHUNK_T;
  float s = 0.f;
  for (int t = 0; t < CHUNK_T; ++t)
    s += emb[(size_t)xr[t] * D_DIM + d];
  csum[((size_t)(b * NCHUNK + c)) * D_DIM + d] = s;
}

// ---------------------------------------------------------------------------
// Stage 1b: prefix mean -> bf16 hi/lo planes in [M][D] row-major
// ---------------------------------------------------------------------------
__global__ __launch_bounds__(256) void MHLLM_prefix_mean(
    const int* __restrict__ x, const float* __restrict__ emb,
    const float* __restrict__ csum,
    __bf16* __restrict__ hHi, __bf16* __restrict__ hLo)
{
  const int b = blockIdx.x / NCHUNK;
  const int c = blockIdx.x % NCHUNK;
  const int d = blockIdx.y * 256 + threadIdx.x;

  float acc = 0.f;
  for (int cc = 0; cc < c; ++cc)
    acc += csum[((size_t)(b * NCHUNK + cc)) * D_DIM + d];

  const int* xr = x + b * T_DIM + c * CHUNK_T;
  for (int t = 0; t < CHUNK_T; ++t) {
    acc += emb[(size_t)xr[t] * D_DIM + d];
    const int tg = c * CHUNK_T + t;
    const float mean = acc / (float)(tg + 1);
    __bf16 hi, lo; split_f32(mean, hi, lo);
    const size_t o = ((size_t)(b * T_DIM + tg)) * D_DIM + d;
    hHi[o] = hi; hLo[o] = lo;
  }
}

// ---------------------------------------------------------------------------
// Stage 2: W [D][V] fp32 -> Wt hi/lo bf16 [V][D] (LDS-tiled 64x64 transpose)
// ---------------------------------------------------------------------------
__global__ __launch_bounds__(256) void MHLLM_wsplit(
    const float* __restrict__ W, __bf16* __restrict__ wHi, __bf16* __restrict__ wLo)
{
  __shared__ unsigned int tile[64][65];
  const int tid = threadIdx.x;
  const int v0 = blockIdx.x * 64;
  const int d0 = blockIdx.y * 64;

  const int vi = tid & 63;
  const int dq = tid >> 6;
#pragma unroll
  for (int k = 0; k < 16; ++k) {
    const int d = dq * 16 + k;
    const float xv = W[(size_t)(d0 + d) * V_DIM + (v0 + vi)];
    __bf16 hi, lo; split_f32(xv, hi, lo);
    tile[vi][d] = (unsigned int)bf_bits(hi) | ((unsigned int)bf_bits(lo) << 16);
  }
  __syncthreads();

  const int vr = tid >> 2;
  const int ds = (tid & 3) * 16;
  unsigned int* oHi = (unsigned int*)(wHi + (size_t)(v0 + vr) * D_DIM + d0 + ds);
  unsigned int* oLo = (unsigned int*)(wLo + (size_t)(v0 + vr) * D_DIM + d0 + ds);
#pragma unroll
  for (int k = 0; k < 16; k += 2) {
    const unsigned int e0 = tile[vr][ds + k];
    const unsigned int e1 = tile[vr][ds + k + 1];
    oHi[k >> 1] = (e0 & 0xFFFFu) | (e1 << 16);
    oLo[k >> 1] = (e0 >> 16)     | (e1 & 0xFFFF0000u);
  }
}

// ---------------------------------------------------------------------------
// Stage 3: WMMA GEMM  logits = h @ W + b  (bf16x3 emulated fp32)
// block = 8 waves; block tile 128x128; wave tile 32x64 (2x4 subtiles)
// Double-buffered LDS staging via GLOBAL_LOAD_ASYNC_TO_LDS_B128 + ASYNCcnt.
// 24 v_wmma_f32_16x16x32_bf16 per K=32 step per wave.
// ---------------------------------------------------------------------------
__global__ __launch_bounds__(256) void MHLLM_gemm(
    const __bf16* __restrict__ hHi, const __bf16* __restrict__ hLo,
    const __bf16* __restrict__ wHi, const __bf16* __restrict__ wLo,
    const float* __restrict__ bias, float* __restrict__ out)
{
  __shared__ __align__(16) unsigned char smem[2 * BUF_BYTES];   // 81920 B

  const int tid  = threadIdx.x;
  const int lane = tid & 31;
  const int wave = tid >> 5;
  const int l16  = lane & 15;
  const int lhi  = lane >> 4;

  const int mBase = blockIdx.y * BLK_M;        // 64 blocks
  const int nBase = blockIdx.x * BLK_N;        // 250 blocks
  const int wm = wave >> 1;                    // 0..3 -> 32-row band
  const int wn = wave & 1;                     // 0..1 -> 64-col band

  const uint32_t smemBase = (uint32_t)(uintptr_t)smem;   // raw LDS byte offset
  const uint64_t hHi64 = (uint64_t)(uintptr_t)hHi;
  const uint64_t hLo64 = (uint64_t)(uintptr_t)hLo;
  const uint64_t wHi64 = (uint64_t)(uintptr_t)wHi;
  const uint64_t wLo64 = (uint64_t)(uintptr_t)wLo;

  // Stage chunk k0 into buffer `buf` (4 planes x 128 rows x 64B, pitch 80B).
  // 8 async b128 per thread; segment id -> row r = id>>2, 16B segment s = id&3.
  auto stage = [&](int k0, int buf) {
    const uint32_t base = smemBase + (uint32_t)buf * BUF_BYTES;
#pragma unroll
    for (int rep = 0; rep < 2; ++rep) {
      const int id = tid + rep * 256;
      const int r  = id >> 2;
      const int s  = id & 3;
      const uint32_t ldsOff = (uint32_t)(r * LDS_PITCH + s * 16);
      const uint32_t gA = (uint32_t)(((mBase + r) * D_DIM + k0) * 2 + s * 16);
      const uint32_t gB = (uint32_t)(((nBase + r) * D_DIM + k0) * 2 + s * 16);
      async_ld_b128(base + 0 * PLANE_BYTES + ldsOff, hHi64, gA);
      async_ld_b128(base + 1 * PLANE_BYTES + ldsOff, hLo64, gA);
      async_ld_b128(base + 2 * PLANE_BYTES + ldsOff, wHi64, gB);
      async_ld_b128(base + 3 * PLANE_BYTES + ldsOff, wLo64, gB);
    }
  };

  v8f acc[2][4];
#pragma unroll
  for (int i = 0; i < 2; ++i)
#pragma unroll
    for (int j = 0; j < 4; ++j)
#pragma unroll
      for (int e = 0; e < 8; ++e) acc[i][j][e] = 0.0f;

  stage(0, 0);

  for (int kc = 0; kc < NKC; ++kc) {
    wait_asynccnt0();          // my async stores to LDS have landed
    __syncthreads();           // everyone's landed; prior reads of other buf done
    if (kc + 1 < NKC) stage((kc + 1) * KC, (kc + 1) & 1);

    const unsigned char* buf = smem + (kc & 1) * BUF_BYTES;
    const unsigned char* aHiP = buf;
    const unsigned char* aLoP = buf + 1 * PLANE_BYTES;
    const unsigned char* bHiP = buf + 2 * PLANE_BYTES;
    const unsigned char* bLoP = buf + 3 * PLANE_BYTES;

    // A fragments (ISA 16-bit 16x32 layout): lanes 0-15 K 0-7/16-23, 16-31 K 8-15/24-31
    AFrag aHi[2], aLo[2];
#pragma unroll
    for (int i = 0; i < 2; ++i) {
      const uint32_t ab = (uint32_t)((wm * 32 + i * 16 + l16) * LDS_PITCH + lhi * 16);
      aHi[i].h[0] = *(const v8bf*)(aHiP + ab);
      aHi[i].h[1] = *(const v8bf*)(aHiP + ab + 32);
      aLo[i].h[0] = *(const v8bf*)(aLoP + ab);
      aLo[i].h[1] = *(const v8bf*)(aLoP + ab + 32);
    }

#pragma unroll
    for (int j = 0; j < 4; ++j) {
      // B fragment (32x16): lanes 0-15 K 0-7/16-23, lanes 16-31 K 8-15/24-31 (matches A)
      const uint32_t bb = (uint32_t)((wn * 64 + j * 16 + l16) * LDS_PITCH + lhi * 16);
      AFrag bHi, bLo;
      bHi.h[0] = *(const v8bf*)(bHiP + bb);
      bHi.h[1] = *(const v8bf*)(bHiP + bb + 32);
      bLo.h[0] = *(const v8bf*)(bLoP + bb);
      bLo.h[1] = *(const v8bf*)(bLoP + bb + 32);
#pragma unroll
      for (int i = 0; i < 2; ++i) {
        acc[i][j] = __builtin_amdgcn_wmma_f32_16x16x32_bf16(
            false, aHi[i].v, false, bHi.v, (short)0, acc[i][j], false, false);
        acc[i][j] = __builtin_amdgcn_wmma_f32_16x16x32_bf16(
            false, aHi[i].v, false, bLo.v, (short)0, acc[i][j], false, false);
        acc[i][j] = __builtin_amdgcn_wmma_f32_16x16x32_bf16(
            false, aLo[i].v, false, bHi.v, (short)0, acc[i][j], false, false);
      }
    }
  }

  // Epilogue: C/D layout — VGPR v: M = v (lanes 0-15) / v+8 (lanes 16-31), N = lane&15.
  // Nontemporal: 1 GB of logits won't be re-hit in L2.
#pragma unroll
  for (int j = 0; j < 4; ++j) {
    const int n = nBase + wn * 64 + j * 16 + l16;
    const float bv = bias[n];
#pragma unroll
    for (int i = 0; i < 2; ++i) {
      const int m0 = mBase + wm * 32 + i * 16 + lhi * 8;
#pragma unroll
      for (int v = 0; v < 8; ++v)
        __builtin_nontemporal_store(acc[i][j][v] + bv,
                                    out + (size_t)(m0 + v) * V_DIM + n);
    }
  }
}

// ---------------------------------------------------------------------------
// Stage 4: per-row NLL via online softmax (block per row)
// ---------------------------------------------------------------------------
__global__ __launch_bounds__(256) void MHLLM_row_nll(
    const float* __restrict__ logits, const int* __restrict__ y,
    float* __restrict__ nll)
{
  const int row = blockIdx.x;
  const int tid = threadIdx.x;
  const float* lr = logits + (size_t)row * V_DIM;

  float m = -3.402823466e38f, s = 0.f;
  for (int v = tid; v < V_DIM; v += 256) {
    const float l = __builtin_nontemporal_load(lr + v);
    if (l > m) { s *= expf(m - l); m = l; }
    s += expf(l - m);
  }

  __shared__ float sm[256], ss[256];
  sm[tid] = m; ss[tid] = s;
  __syncthreads();
  for (int w = 128; w > 0; w >>= 1) {
    if (tid < w) {
      const float m2 = sm[tid + w], s2 = ss[tid + w];
      const float M  = fmaxf(sm[tid], m2);
      ss[tid] = ss[tid] * expf(sm[tid] - M) + s2 * expf(m2 - M);
      sm[tid] = M;
    }
    __syncthreads();
  }
  if (tid == 0) {
    const float lse = sm[0] + logf(ss[0]);
    nll[row] = lse - lr[y[row]];
  }
}

__global__ __launch_bounds__(256) void MHLLM_loss_reduce(
    const float* __restrict__ nll, float* __restrict__ out_loss)
{
  __shared__ float sb[256];
  float s = 0.f;
  for (int i = threadIdx.x; i < M_DIM; i += 256) s += nll[i];
  sb[threadIdx.x] = s;
  __syncthreads();
  for (int w = 128; w > 0; w >>= 1) {
    if (threadIdx.x < w) sb[threadIdx.x] += sb[threadIdx.x + w];
    __syncthreads();
  }
  if (threadIdx.x == 0) *out_loss = sb[0] / (float)M_DIM;
}

// ---------------------------------------------------------------------------
// Workspace layout (bytes):
//   hHi  @ 0           : M*D*2   = 16,777,216
//   hLo  @ 16,777,216  : M*D*2   = 16,777,216
//   wHi  @ 33,554,432  : V*D*2   = 65,536,000
//   wLo  @ 99,090,432  : V*D*2   = 65,536,000
//   nll  @ 164,626,432 : M*4     = 32,768
//   csum @ 164,659,200 : B*NC*D*4= 262,144      (total ~157.3 MB)
// ---------------------------------------------------------------------------
extern "C" void kernel_launch(void* const* d_in, const int* in_sizes, int n_in,
                              void* d_out, int out_size, void* d_ws, size_t ws_size,
                              hipStream_t stream) {
  (void)in_sizes; (void)n_in; (void)out_size; (void)ws_size;

  const int*   x    = (const int*)  d_in[0];
  const int*   y    = (const int*)  d_in[1];
  const float* emb  = (const float*)d_in[2];
  const float* W    = (const float*)d_in[3];
  const float* bias = (const float*)d_in[4];
  float* out = (float*)d_out;

  char* ws = (char*)d_ws;
  __bf16* hHi = (__bf16*)(ws + 0);
  __bf16* hLo = (__bf16*)(ws + 16777216ULL);
  __bf16* wHi = (__bf16*)(ws + 33554432ULL);
  __bf16* wLo = (__bf16*)(ws + 99090432ULL);
  float*  nll = (float*) (ws + 164626432ULL);
  float*  csm = (float*) (ws + 164659200ULL);

  MHLLM_chunk_sums <<<dim3(B_DIM * NCHUNK, D_DIM / 256), 256, 0, stream>>>(x, emb, csm);
  MHLLM_prefix_mean<<<dim3(B_DIM * NCHUNK, D_DIM / 256), 256, 0, stream>>>(x, emb, csm, hHi, hLo);
  MHLLM_wsplit     <<<dim3(V_DIM / 64, D_DIM / 64), 256, 0, stream>>>(W, wHi, wLo);
  MHLLM_gemm       <<<dim3(V_DIM / BLK_N, M_DIM / BLK_M), 256, 0, stream>>>(hHi, hLo, wHi, wLo, bias, out);
  MHLLM_row_nll    <<<dim3(M_DIM), 256, 0, stream>>>(out, y, nll);
  MHLLM_loss_reduce<<<dim3(1), 256, 0, stream>>>(nll, out + (size_t)M_DIM * V_DIM);
}
